// GAE_45775761441312
// MI455X (gfx1250) — matrix operand order repaired and checked by reading, not compiled, not run on hardware
//
#include <hip/hip_runtime.h>

// ---------------------------------------------------------------------------
// Problem constants (from the reference)
// ---------------------------------------------------------------------------
#define NNODES 90000
#define NEDGES 540000
#define NGRAPH 10000   // NNODES / 9
#define MTILES 5625    // NNODES / 16

typedef __attribute__((ext_vector_type(8)))  __bf16 bf16x8;
typedef __attribute__((ext_vector_type(16))) __bf16 bf16x16;
typedef __attribute__((ext_vector_type(8)))  float  v8f;

static __device__ __forceinline__ unsigned short f2bf(float f) {
    unsigned int u = __float_as_uint(f);
    unsigned int r = (u + 0x7FFFu + ((u >> 16) & 1u)) >> 16;   // round-to-nearest-even
    return (unsigned short)r;
}
static __device__ __forceinline__ float bf2f(unsigned short h) {
    return __uint_as_float(((unsigned int)h) << 16);
}
static __device__ __forceinline__ bf16x16 cat8(bf16x8 lo, bf16x8 hi) {
    return __builtin_shufflevector(lo, hi, 0,1,2,3,4,5,6,7,8,9,10,11,12,13,14,15);
}

// ---------------------------------------------------------------------------
// Weight packing: w[K x Nout] f32 (row-major) -> wp[Nout x Kpad] bf16
// (per-output-column contiguous K so WMMA B-fragment loads are b128s)
// ---------------------------------------------------------------------------
__global__ void k_pack_w(const float* __restrict__ w, unsigned short* __restrict__ wp,
                         int K, int Nout, int Kpad) {
    int i = blockIdx.x * blockDim.x + threadIdx.x;
    if (i >= Nout * Kpad) return;
    int n = i / Kpad, k = i % Kpad;
    float v = (k < K) ? w[(size_t)k * Nout + n] : 0.0f;
    wp[i] = f2bf(v);
}

__global__ void k_copy_f32(const float* __restrict__ src, float* __restrict__ dst, int n) {
    int i = blockIdx.x * blockDim.x + threadIdx.x;
    if (i < n) dst[i] = src[i];
}

// Segment-sum of x (30 feats) into h0f (self term pre-initialized with x).
__global__ void k_agg_x(const float* __restrict__ x, const int* __restrict__ src,
                        const int* __restrict__ dst, float* __restrict__ h0f) {
    int e = blockIdx.x * 8 + (threadIdx.x >> 5);
    int f = threadIdx.x & 31;
    if (e >= NEDGES || f >= 30) return;
    int s = src[e], d = dst[e];
    unsafeAtomicAdd(&h0f[(size_t)d * 30 + f], x[(size_t)s * 30 + f]);
}

// h0f (N x 30 f32) -> h0bf (N x 32 bf16, zero padded K)
__global__ void k_build_h0bf(const float* __restrict__ h0f, unsigned short* __restrict__ h0bf) {
    int i = blockIdx.x * blockDim.x + threadIdx.x;
    if (i >= NNODES * 32) return;
    int nd = i >> 5, f = i & 31;
    float v = (f < 30) ? h0f[(size_t)nd * 30 + f] : 0.0f;
    h0bf[i] = f2bf(v);
}

// Segment-sum of 128-wide bf16 features into f32 accumulator (self term pre-init).
__global__ void k_agg_z(const unsigned short* __restrict__ zbf, const int* __restrict__ src,
                        const int* __restrict__ dst, float* __restrict__ hf) {
    int e = blockIdx.x * 2 + (threadIdx.x >> 7);
    int f = threadIdx.x & 127;
    if (e >= NEDGES) return;
    int s = src[e], d = dst[e];
    unsafeAtomicAdd(&hf[(size_t)d * 128 + f], bf2f(zbf[(size_t)s * 128 + f]));
}

__global__ void k_f32_to_bf16(const float* __restrict__ in, unsigned short* __restrict__ out, int n) {
    int i = blockIdx.x * blockDim.x + threadIdx.x;
    if (i < n) out[i] = f2bf(in[i]);
}

// ---------------------------------------------------------------------------
// Wave-level bf16 WMMA GEMM, full-width strip version:
//   one wave owns a 16-row strip of A and produces ALL NT*16 output columns.
//   A fragment is loaded ONCE per K-step and reused by NT WMMAs (weights are
//   tiny and L0-resident, so the extra B loads are cache hits).
//
//   A : [NNODES x KPAD] bf16 row-major activations
//   Wp: [NOUT  x KPAD] bf16 packed weights (column n contiguous in K)
//
// Fragment layouts per ISA 7.12.2 (wave32):
//   A 16x32: lanes 0-15 (row m) hold K kb+0..7 / kb+16..23,
//            lanes 16-31 hold K kb+8..15 / kb+24..31.
//   B 32x16: lane = output column, same K split as A (kb+8*half+0..7 / kb+16+8*half+0..7).
//   C/D    : VGPR r -> rows half*8+r, column = lane&15.
//
// WRITEF / WRITEB / RELU are compile-time so the epilogue is branch-free and
// the 8 row-stores per tile use immediate offsets from one base pointer
// (fixed (t,r) store: lanes 0-15 and 16-31 each cover a contiguous 64B chunk).
// blockDim = 32 (single wave); blockIdx.x = row tile.
// ---------------------------------------------------------------------------
template <int KPAD, int NT, bool WRITEF, bool WRITEB, bool RELU>
__global__ void k_gemm_wmma(const unsigned short* __restrict__ A,
                            const unsigned short* __restrict__ Wp,
                            const float* __restrict__ bias,
                            float* __restrict__ outF,
                            unsigned short* __restrict__ outB) {
    constexpr int NOUT = NT * 16;
    const int lane = threadIdx.x & 31;
    const int half = lane >> 4;
    const int m    = lane & 15;
    const int row  = blockIdx.x * 16 + m;
    const unsigned short* arow = A + (size_t)row * KPAD;

    v8f c[NT];
#pragma unroll
    for (int t = 0; t < NT; ++t) c[t] = (v8f){0.f, 0.f, 0.f, 0.f, 0.f, 0.f, 0.f, 0.f};

#pragma unroll
    for (int kb = 0; kb < KPAD; kb += 32) {
        bf16x8 a0 = *reinterpret_cast<const bf16x8*>(arow + kb + half * 8);
        bf16x8 a1 = *reinterpret_cast<const bf16x8*>(arow + kb + 16 + half * 8);
        bf16x16 av = cat8(a0, a1);
#pragma unroll
        for (int t = 0; t < NT; ++t) {
            const unsigned short* bcol =
                Wp + (size_t)(t * 16 + m) * KPAD + kb + half * 8;
            bf16x8 b0 = *reinterpret_cast<const bf16x8*>(bcol);
            bf16x8 b1 = *reinterpret_cast<const bf16x8*>(bcol + 16);
            bf16x16 bv = cat8(b0, b1);
            c[t] = __builtin_amdgcn_wmma_f32_16x16x32_bf16(false, av, false, bv,
                                                           (short)0, c[t], false, false);
        }
    }

    const size_t base = (size_t)(blockIdx.x * 16 + half * 8) * NOUT + m;
#pragma unroll
    for (int t = 0; t < NT; ++t) {
        const float bsc = bias[t * 16 + m];
        float*          pf = WRITEF ? (outF + base + t * 16) : nullptr;
        unsigned short* pb = WRITEB ? (outB + base + t * 16) : nullptr;
#pragma unroll
        for (int r = 0; r < 8; ++r) {
            float v = c[t][r] + bsc;
            if constexpr (RELU) v = fmaxf(v, 0.0f);
            if constexpr (WRITEF) pf[(size_t)r * NOUT] = v;
            if constexpr (WRITEB) pb[(size_t)r * NOUT] = f2bf(v);
        }
    }
}

// ---------------------------------------------------------------------------
// Per-graph 9x9 Gram matrix: adj[g,i,j] = dot(z2s[g*9+i], z2t[g*9+j]) over 64.
// ---------------------------------------------------------------------------
__global__ void k_adj(const float* __restrict__ z2s, const float* __restrict__ z2t,
                      float* __restrict__ out) {
    __shared__ float ss[9 * 64];
    __shared__ float tt[9 * 64];
    int g = blockIdx.x;
    for (int i = threadIdx.x; i < 9 * 64; i += blockDim.x) {
        ss[i] = z2s[(size_t)g * 576 + i];
        tt[i] = z2t[(size_t)g * 576 + i];
    }
    __syncthreads();
    for (int t = threadIdx.x; t < 81; t += blockDim.x) {
        int i = t / 9, j = t % 9;
        float acc = 0.f;
#pragma unroll
        for (int k = 0; k < 64; ++k) acc += ss[i * 64 + k] * tt[j * 64 + k];
        out[(size_t)g * 81 + t] = acc;
    }
}

// ---------------------------------------------------------------------------
extern "C" void kernel_launch(void* const* d_in, const int* in_sizes, int n_in,
                              void* d_out, int out_size, void* d_ws, size_t ws_size,
                              hipStream_t stream) {
    const float* x   = (const float*)d_in[0];
    const int*   ei  = (const int*)d_in[1];
    const int*   src = ei;
    const int*   dst = ei + NEDGES;

    // weights / biases in setup order: (w1,b1..w4,b4) for 's' then 't'
    const float* W[8];
    const float* B[8];
    for (int i = 0; i < 8; ++i) {
        W[i] = (const float*)d_in[2 + 2 * i];
        B[i] = (const float*)d_in[3 + 2 * i];
    }
    const int Ks[4] = {30, 128, 128, 128};
    const int Kp[4] = {32, 128, 128, 128};
    const int No[4] = {128, 128, 128, 64};

    // ---- workspace layout (256B aligned) ----
    char*  ws  = (char*)d_ws;
    size_t off = 0;
    auto alloc = [&](size_t bytes) -> size_t {
        off = (off + 255) & ~(size_t)255;
        size_t o = off;
        off += bytes;
        return o;
    };
    unsigned short* wp[8];
    for (int br = 0; br < 2; ++br)
        for (int l = 0; l < 4; ++l) {
            int i = br * 4 + l;
            wp[i] = (unsigned short*)(ws + alloc((size_t)No[l] * Kp[l] * 2));
        }
    unsigned short* h0bf = (unsigned short*)(ws + alloc((size_t)NNODES * 32 * 2));
    unsigned short* abf  = (unsigned short*)(ws + alloc((size_t)NNODES * 128 * 2));
    unsigned short* z1bf = (unsigned short*)(ws + alloc((size_t)NNODES * 128 * 2));
    float*          h2f  = (float*)(ws + alloc((size_t)NNODES * 128 * 4));
    float*          h0f  = h2f; // alias: h0f (N x 30 f32) only lives before h2f is used
    float*          z2s  = (float*)(ws + alloc((size_t)NNODES * 64 * 4));
    float*          z2t  = (float*)(ws + alloc((size_t)NNODES * 64 * 4));
    (void)ws_size; (void)in_sizes; (void)n_in; (void)out_size;

    // ---- pack weights to bf16 fragment layout ----
    for (int i = 0; i < 8; ++i) {
        int l = i & 3;
        int total = No[l] * Kp[l];
        k_pack_w<<<(total + 255) / 256, 256, 0, stream>>>(W[i], wp[i], Ks[l], No[l], Kp[l]);
    }

    // ---- h0 = x + segment_sum(x[src] -> dst), converted to bf16 (K padded to 32)
    k_copy_f32<<<(NNODES * 30 + 255) / 256, 256, 0, stream>>>(x, h0f, NNODES * 30);
    k_agg_x<<<(NEDGES + 7) / 8, 256, 0, stream>>>(x, src, dst, h0f);
    k_build_h0bf<<<(NNODES * 32 + 255) / 256, 256, 0, stream>>>(h0f, h0bf);

    // ---- two branches: GIN1 -> GIN2 ----
    for (int br = 0; br < 2; ++br) {
        int b0 = br * 4;
        float* z2 = br ? z2t : z2s;

        // layer 1: a1 = relu(h0 @ w1 + b1)           [N x 128] bf16
        k_gemm_wmma<32, 8, false, true, true>
            <<<MTILES, 32, 0, stream>>>(h0bf, wp[b0 + 0], B[b0 + 0], nullptr, abf);
        // layer 2: z1 = a1 @ w2 + b2 ; f32 (self term of next agg) + bf16 (gather src)
        k_gemm_wmma<128, 8, true, true, false>
            <<<MTILES, 32, 0, stream>>>(abf, wp[b0 + 1], B[b0 + 1], h2f, z1bf);
        // h2 = z1 + segment_sum(z1[src] -> dst)
        k_agg_z<<<(NEDGES + 1) / 2, 256, 0, stream>>>(z1bf, src, dst, h2f);
        k_f32_to_bf16<<<(NNODES * 128 + 255) / 256, 256, 0, stream>>>(h2f, z1bf, NNODES * 128);
        // layer 3: a2 = relu(h2 @ w3 + b3)
        k_gemm_wmma<128, 8, false, true, true>
            <<<MTILES, 32, 0, stream>>>(z1bf, wp[b0 + 2], B[b0 + 2], nullptr, abf);
        // layer 4: z2 = a2 @ w4 + b4                 [N x 64] f32
        k_gemm_wmma<128, 4, true, false, false>
            <<<MTILES, 32, 0, stream>>>(abf, wp[b0 + 3], B[b0 + 3], z2, nullptr);
    }

    // ---- per-graph 9x9 Gram matrices -> output [N x 9] f32 ----
    k_adj<<<NGRAPH, 128, 0, stream>>>(z2s, z2t, (float*)d_out);
}
